// PwcNet_Single_75522704933024
// MI455X (gfx1250) — matrix operand order repaired and checked by reading, not compiled, not run on hardware
//
#include <hip/hip_runtime.h>
#include <hip/hip_bf16.h>
#include <stdint.h>

typedef __attribute__((ext_vector_type(16))) _Float16 v16h;
typedef __attribute__((ext_vector_type(8)))  _Float16 v8h;
typedef __attribute__((ext_vector_type(8)))  float    v8f;
typedef __attribute__((ext_vector_type(4)))  unsigned int v4u;

#define EL_THREADS 256
#define CONV_WAVES 4
static inline int eblocks(size_t n) { return (int)((n + EL_THREADS - 1) / EL_THREADS); }

// ---------------------------------------------------------------------------
// Elementwise / glue kernels
// ---------------------------------------------------------------------------
__global__ void k_zero_h(_Float16* p, size_t n) {
  size_t i = (size_t)blockIdx.x * blockDim.x + threadIdx.x;
  if (i < n) p[i] = (_Float16)0.f;
}

__global__ void k_pack_f32(const float* __restrict__ src, int Csrc,
                           _Float16* __restrict__ dst, int Cdst, int coff, size_t n) {
  size_t i = (size_t)blockIdx.x * blockDim.x + threadIdx.x;
  if (i >= n) return;
  int c = (int)(i % Csrc);
  size_t p = i / Csrc;
  dst[p * (size_t)Cdst + coff + c] = (_Float16)src[i];
}

__global__ void k_pack_f16(const _Float16* __restrict__ src, int Csrc,
                           _Float16* __restrict__ dst, int Cdst, int coff, size_t n) {
  size_t i = (size_t)blockIdx.x * blockDim.x + threadIdx.x;
  if (i >= n) return;
  int c = (int)(i % Csrc);
  size_t p = i / Csrc;
  dst[p * (size_t)Cdst + coff + c] = src[i];
}

__global__ void k_relu_add(const float* __restrict__ a, const float* __restrict__ b,
                           float* __restrict__ o, size_t n) {
  size_t i = (size_t)blockIdx.x * blockDim.x + threadIdx.x;
  if (i >= n) return;
  float v = a[i] + (b ? b[i] : 0.f);
  o[i] = v > 0.f ? v : 0.f;
}

__global__ void k_scale(const float* __restrict__ a, float* __restrict__ o, float s, size_t n) {
  size_t i = (size_t)blockIdx.x * blockDim.x + threadIdx.x;
  if (i < n) o[i] = a[i] * s;
}

// Bilinear 2x upsample of a 1-channel map, fused with the *2.0 flow scale.
__global__ void k_up2x(const float* __restrict__ src, float* __restrict__ dst,
                       int B, int Hs, int Ws) {
  int Hd = Hs * 2, Wd = Ws * 2;
  size_t n = (size_t)B * Hd * Wd;
  size_t i = (size_t)blockIdx.x * blockDim.x + threadIdx.x;
  if (i >= n) return;
  int x = (int)(i % Wd);
  size_t r = i / Wd;
  int y = (int)(r % Hd);
  int b = (int)(r / Hd);
  float fy = (y + 0.5f) * 0.5f - 0.5f;
  float fx = (x + 0.5f) * 0.5f - 0.5f;
  float y0f = floorf(fy), x0f = floorf(fx);
  float ay = fy - y0f, ax = fx - x0f;
  int y0 = (int)y0f, x0 = (int)x0f;
  int y0c = y0 < 0 ? 0 : (y0 > Hs - 1 ? Hs - 1 : y0);
  int y1c = (y0 + 1) < 0 ? 0 : ((y0 + 1) > Hs - 1 ? Hs - 1 : (y0 + 1));
  int x0c = x0 < 0 ? 0 : (x0 > Ws - 1 ? Ws - 1 : x0);
  int x1c = (x0 + 1) < 0 ? 0 : ((x0 + 1) > Ws - 1 ? Ws - 1 : (x0 + 1));
  const float* sp = src + (size_t)b * Hs * Ws;
  float v00 = sp[(size_t)y0c * Ws + x0c];
  float v01 = sp[(size_t)y0c * Ws + x1c];
  float v10 = sp[(size_t)y1c * Ws + x0c];
  float v11 = sp[(size_t)y1c * Ws + x1c];
  float top = v00 + ax * (v01 - v00);
  float bot = v10 + ax * (v11 - v10);
  dst[i] = 2.0f * (top + ay * (bot - top));
}

// Horizontal-only inverse warp (flow dy == 0 identically in this network).
__global__ void k_warp_x(const float* __restrict__ img, const float* __restrict__ dx,
                         float* __restrict__ out, int B, int H, int W, int C) {
  size_t n = (size_t)B * H * W * C;
  size_t i = (size_t)blockIdx.x * blockDim.x + threadIdx.x;
  if (i >= n) return;
  int c = (int)(i % C);
  size_t p = i / C;
  int x = (int)(p % W);
  size_t r = p / W;
  int y = (int)(r % H);
  int b = (int)(r / H);
  float qx = (float)x + dx[p];
  float x0f = floorf(qx);
  x0f = x0f < 0.f ? 0.f : (x0f > (float)(W - 2) ? (float)(W - 2) : x0f);
  float ax = qx - x0f;
  ax = ax < 0.f ? 0.f : (ax > 1.f ? 1.f : ax);
  int x0 = (int)x0f;
  const float* row = img + (((size_t)b * H + y) * (size_t)W) * C;
  float l = row[(size_t)x0 * C + c];
  float rr = row[(size_t)(x0 + 1) * C + c];
  out[i] = l + ax * (rr - l);
}

// Horizontal cost volume (radius 4 -> 9 channels), writes f16 into packed input.
__global__ void k_costvol(const float* __restrict__ f1, const float* __restrict__ f2,
                          _Float16* __restrict__ dst, int Cdst,
                          int B, int H, int W, int C) {
  size_t n = (size_t)B * H * W * 9;
  size_t i = (size_t)blockIdx.x * blockDim.x + threadIdx.x;
  if (i >= n) return;
  int j = (int)(i % 9);
  size_t p = i / 9;                  // p = ((b*H + y)*W + x)
  int x = (int)(p % W);
  int xs = x + j - 4;
  float s = 0.f;
  if (xs >= 0 && xs < W) {
    const float* a = f1 + p * (size_t)C;
    const float* bb = f2 + (p + (size_t)(xs - x)) * (size_t)C;  // same row shift
    for (int c = 0; c < C; ++c) s += a[c] * bb[c];
    s *= (1.0f / (float)C);
  }
  dst[p * (size_t)Cdst + j] = (_Float16)s;
}

// ---------------------------------------------------------------------------
// Weight repack: HWIO f32 (3,3,Cin,Cout) -> WMMA B-fragment f16 layout
//   wpk[t][kc][nc][lane(N=lane&15)][i(K=(lane>>4)*8+(i&7)+(i>>3)*16)], zero-padded.
// ---------------------------------------------------------------------------
__global__ void k_repack_w(const float* __restrict__ w, _Float16* __restrict__ wpk,
                           int Cin, int Cout, int Kch, int Nch, size_t total) {
  size_t idx = (size_t)blockIdx.x * blockDim.x + threadIdx.x;
  if (idx >= total) return;
  int i = (int)(idx & 15);
  int lane = (int)((idx >> 4) & 31);
  size_t r = idx >> 9;
  int nc = (int)(r % Nch); r /= Nch;
  int kc = (int)(r % Kch);
  int t  = (int)(r / Kch);
  int k = kc * 32 + (lane >> 4) * 8 + (i & 7) + (i >> 3) * 16;
  int n = nc * 16 + (lane & 15);
  _Float16 v = (_Float16)0.f;
  if (k < Cin && n < Cout) v = (_Float16)w[((size_t)t * Cin + k) * Cout + n];
  wpk[idx] = v;
}

// ---------------------------------------------------------------------------
// Implicit-GEMM 3x3 conv (SAME, optional dilation) on v_wmma_f32_16x16x32_f16.
// Block = 4 waves; each wave owns 16 pixels (along W) x 16 output channels with
// a shared nc, so B fragments are reused out of L0 across the 4 waves.
// A loads are branchless: addresses clamped in-range, then cndmask-zeroed.
// A layout (ISA 16-bit 16x32): lane m=lane&15 is row M; halves 0..7 hold
// K = kbase + (lane<16?0:8) + 0..7, halves 8..15 hold K = that + 16.
// C/D: element r -> pixel x0 + r + (lane>=16)*8, channel nbase + (lane&15).
// ---------------------------------------------------------------------------
static __device__ inline v16h a_frag_masked(const _Float16* p, bool ok) {
  union U { v4u u; v8h h; };
  U u0, u1;
  u0.u = *(const v4u*)p;          // always-safe (clamped) addresses
  u1.u = *(const v4u*)(p + 16);
  v4u z = {0u, 0u, 0u, 0u};
  u0.u = ok ? u0.u : z;           // v_cndmask, no branch
  u1.u = ok ? u1.u : z;
  v16h a;
#pragma unroll
  for (int q = 0; q < 8; ++q) { a[q] = u0.h[q]; a[8 + q] = u1.h[q]; }
  return a;
}

__global__ __launch_bounds__(32 * CONV_WAVES) void k_conv3x3_wmma(
    const _Float16* __restrict__ in0, int C0,
    const _Float16* __restrict__ in1, int C1,
    const _Float16* __restrict__ wpk,
    const float* __restrict__ bias,
    _Float16* __restrict__ out, int CoutTot,
    float* __restrict__ outF32,
    int B, int H, int W, int CoutReal, int Nch, int dil, int act) {
  const int lane = (int)threadIdx.x & 31;
  const int wave = (int)threadIdx.x >> 5;
  const int x0 = (blockIdx.x * CONV_WAVES + wave) * 16;
  const int y = blockIdx.y % H;
  const int b = blockIdx.y / H;
  const int nc = blockIdx.z;
  const int laneHi = lane >> 4;
  const int m = lane & 15;
  const int px = x0 + m;
  const int Kch0 = C0 >> 5;
  const int Kch1 = C1 >> 5;
  const int Kch = Kch0 + Kch1;
  const size_t wstep = (size_t)Nch * 512;   // elements between consecutive kc fragments

  v8f acc = {};
#pragma unroll
  for (int ky = 0; ky < 3; ++ky) {
    const int iy = y + (ky - 1) * dil;
    const bool yok = ((unsigned)iy < (unsigned)H);
    const int iyc = iy < 0 ? 0 : (iy >= H ? H - 1 : iy);
#pragma unroll
    for (int kx = 0; kx < 3; ++kx) {
      const int ix = px + (kx - 1) * dil;
      const bool ok = yok && ((unsigned)ix < (unsigned)W);
      const int ixc = ix < 0 ? 0 : (ix >= W ? W - 1 : ix);
      const size_t pix = ((size_t)b * H + iyc) * W + ixc;
      const int t = ky * 3 + kx;
      const _Float16* wq = wpk + ((size_t)t * Kch * Nch + nc) * 512 + (size_t)lane * 16;
      const _Float16* pa = in0 + pix * (size_t)C0 + laneHi * 8;
      for (int kc = 0; kc < Kch0; ++kc) {
        v16h a = a_frag_masked(pa, ok);
        v16h bf = *(const v16h*)wq;
        acc = __builtin_amdgcn_wmma_f32_16x16x32_f16(false, a, false, bf,
                                                     (short)0, acc, false, false);
        pa += 32;
        wq += wstep;
      }
      if (Kch1 > 0) {
        const _Float16* pb = in1 + pix * (size_t)C1 + laneHi * 8;
        for (int kc = 0; kc < Kch1; ++kc) {
          v16h a = a_frag_masked(pb, ok);
          v16h bf = *(const v16h*)wq;
          acc = __builtin_amdgcn_wmma_f32_16x16x32_f16(false, a, false, bf,
                                                       (short)0, acc, false, false);
          pb += 32;
          wq += wstep;
        }
      }
    }
  }

  const int oc = nc * 16 + m;
#pragma unroll
  for (int r = 0; r < 8; ++r) {
    const int ox = x0 + r + laneHi * 8;
    if (ox < W && oc < CoutReal) {
      float v = acc[r] + bias[oc];
      if (act) v = (v > 0.f) ? v : 0.1f * v;
      size_t pix = ((size_t)b * H + y) * W + ox;
      if (outF32) { if (oc == 0) outF32[pix] = v; }
      else out[pix * (size_t)CoutTot + oc] = (_Float16)v;
    }
  }
}

// ---------------------------------------------------------------------------
// Host orchestration
// ---------------------------------------------------------------------------
extern "C" void kernel_launch(void* const* d_in, const int* in_sizes, int n_in,
                              void* d_out, int out_size, void* d_ws, size_t ws_size,
                              hipStream_t stream) {
  if (n_in < 84) return;
  const int B = 8;
  const int Hs[5] = {64, 32, 16, 8, 4};
  const int Wsz[5] = {208, 104, 52, 26, 13};
  const int Cs[5] = {32, 64, 96, 128, 196};
  size_t Ps[5];
  for (int l = 0; l < 5; ++l) Ps[l] = (size_t)B * Hs[l] * Wsz[l];
  const int CinP0[5] = {64, 96, 128, 160, 32};  // padded first-conv Cin
  const int cin0[5] = {42, 74, 106, 139, 9};    // real first-conv Cin

  // ---- feature input pointers (detect interleaved vs sorted flattening) ----
  const float *f1[5], *f2[5];
  bool interleaved = (in_sizes[1] == in_sizes[0]);
  for (int l = 0; l < 5; ++l) {
    if (interleaved) { f1[l] = (const float*)d_in[2 * l]; f2[l] = (const float*)d_in[2 * l + 1]; }
    else             { f1[l] = (const float*)d_in[l];     f2[l] = (const float*)d_in[5 + l]; }
  }

  // ---- param leaves (detect w/b order and cn-first vs dec6-first) ----
  struct ConvLayer { const float* w; const float* b; int cin; int cout; };
  ConvLayer dec[5][6];
  ConvLayer cn[7];
  bool bias_first = (in_sizes[10] <= 512);
  int wslot = bias_first ? 11 : 10;
  bool cn_first = (in_sizes[wslot] > 20000);  // cn L0 w = 38016 vs dec6 L0 w = 10368
  int pi = 10;
  auto take = [&](ConvLayer& L, int ci, int co) {
    if (bias_first) { L.b = (const float*)d_in[pi]; L.w = (const float*)d_in[pi + 1]; }
    else            { L.w = (const float*)d_in[pi]; L.b = (const float*)d_in[pi + 1]; }
    pi += 2; L.cin = ci; L.cout = co;
  };
  const int decOuts[6] = {128, 128, 96, 64, 32, 1};
  auto takeDec = [&](int l) {
    int ins[6] = {cin0[l], 128, 256, 224, 160, 96};
    for (int j = 0; j < 6; ++j) take(dec[l][j], ins[j], decOuts[j]);
  };
  const int cnIns[7] = {33, 128, 128, 128, 96, 64, 32};
  const int cnOuts[7] = {128, 128, 128, 96, 64, 32, 1};
  auto takeCn = [&]() { for (int j = 0; j < 7; ++j) take(cn[j], cnIns[j], cnOuts[j]); };
  if (cn_first) { takeCn(); for (int l = 0; l < 5; ++l) takeDec(l); }
  else          { for (int l = 4; l >= 0; --l) takeDec(l); takeCn(); }

  // ---- workspace layout ----
  char* base = (char*)d_ws;
  size_t off = 0;
  auto alloc = [&](size_t bytes) -> char* {
    off = (off + 255) & ~(size_t)255;
    char* p = base + off; off += bytes; return p;
  };
  size_t P2 = Ps[0];
  _Float16* o1 = (_Float16*)alloc(P2 * 128 * 2);
  _Float16* o2 = (_Float16*)alloc(P2 * 128 * 2);
  _Float16* o3 = (_Float16*)alloc(P2 * 96 * 2);
  _Float16* o4 = (_Float16*)alloc(P2 * 64 * 2);
  _Float16* o5 = (_Float16*)alloc(P2 * 32 * 2);
  _Float16* packedIn = (_Float16*)alloc(P2 * 64 * 2);   // largest packed input (L2 / cn)
  float* warped = (float*)alloc(P2 * 32 * 4);           // largest warped feature map
  _Float16* wpk = (_Float16*)alloc((size_t)9 * 8 * 8 * 512 * 2);  // weight fragment scratch
  float* fxBuf = (float*)alloc(P2 * 4);
  float* cnOut = (float*)alloc(P2 * 4);
  float* flow2raw = (float*)alloc(P2 * 4);
  float* flow2 = (float*)alloc(P2 * 4);
  float* flowB[5];
  for (int l = 0; l < 5; ++l) flowB[l] = (float*)alloc(Ps[l] * 4);
  float* upB[4];
  for (int l = 0; l < 4; ++l) upB[l] = (float*)alloc(Ps[l] * 4);
  if (off > ws_size) return;

  // ---- conv launcher: repack weights then WMMA conv ----
  auto conv = [&](const _Float16* in0, int C0, const _Float16* in1, int C1,
                  const ConvLayer& L, _Float16* out, int CoutTot, float* outF32,
                  int H, int W, int dil, int act) {
    int Kch = (C0 + C1) >> 5;
    int Nch = (L.cout + 15) >> 4;
    size_t tot = (size_t)9 * Kch * Nch * 512;
    k_repack_w<<<eblocks(tot), EL_THREADS, 0, stream>>>(L.w, wpk, L.cin, L.cout, Kch, Nch, tot);
    int xTiles = (W + 15) / 16;
    dim3 grid((xTiles + CONV_WAVES - 1) / CONV_WAVES, B * H, Nch);
    k_conv3x3_wmma<<<grid, dim3(32 * CONV_WAVES), 0, stream>>>(
        in0, C0, in1, C1, wpk, L.b, out, CoutTot, outF32, B, H, W, L.cout, Nch, dil, act);
  };

  auto decoder = [&](const ConvLayer* lay, const _Float16* xin, int CP, int H, int W) {
    conv(xin, CP, nullptr, 0, lay[0], o1, 128, nullptr, H, W, 1, 1);
    conv(o1, 128, nullptr, 0, lay[1], o2, 128, nullptr, H, W, 1, 1);
    conv(o1, 128, o2, 128, lay[2], o3, 96, nullptr, H, W, 1, 1);
    conv(o2, 128, o3, 96, lay[3], o4, 64, nullptr, H, W, 1, 1);
    conv(o3, 96, o4, 64, lay[4], o5, 32, nullptr, H, W, 1, 1);
    conv(o4, 64, o5, 32, lay[5], nullptr, 0, fxBuf, H, W, 1, 0);
  };

  // ---- level 6 ----
  {
    int l = 4; size_t P = Ps[l]; int H = Hs[l], W = Wsz[l], C = Cs[l], CP = CinP0[l];
    k_zero_h<<<eblocks(P * CP), EL_THREADS, 0, stream>>>(packedIn, P * CP);
    k_costvol<<<eblocks(P * 9), EL_THREADS, 0, stream>>>(f1[l], f2[l], packedIn, CP, B, H, W, C);
    decoder(dec[l], packedIn, CP, H, W);
    k_relu_add<<<eblocks(P), EL_THREADS, 0, stream>>>(fxBuf, (const float*)nullptr, flowB[l], P);
  }

  // ---- levels 5..2 ----
  for (int l = 3; l >= 0; --l) {
    size_t P = Ps[l]; int H = Hs[l], W = Wsz[l], C = Cs[l], CP = CinP0[l];
    k_up2x<<<eblocks(P), EL_THREADS, 0, stream>>>(flowB[l + 1], upB[l], B, Hs[l + 1], Wsz[l + 1]);
    k_warp_x<<<eblocks(P * (size_t)C), EL_THREADS, 0, stream>>>(f2[l], upB[l], warped, B, H, W, C);
    k_zero_h<<<eblocks(P * CP), EL_THREADS, 0, stream>>>(packedIn, P * CP);
    k_costvol<<<eblocks(P * 9), EL_THREADS, 0, stream>>>(f1[l], warped, packedIn, CP, B, H, W, C);
    k_pack_f32<<<eblocks(P * (size_t)C), EL_THREADS, 0, stream>>>(f1[l], C, packedIn, CP, 9, P * (size_t)C);
    int foff = (l == 3) ? 138 : (9 + C);  // dec5 gets both flow channels (dy channel stays 0)
    k_pack_f32<<<eblocks(P), EL_THREADS, 0, stream>>>(upB[l], 1, packedIn, CP, foff, P);
    decoder(dec[l], packedIn, CP, H, W);
    float* dstFlow = (l == 0) ? flow2raw : flowB[l];
    k_relu_add<<<eblocks(P), EL_THREADS, 0, stream>>>(fxBuf, upB[l], dstFlow, P);
  }

  // ---- context network (dilated), ping-pong on o1/o2 ----
  {
    size_t P = P2; int H = Hs[0], W = Wsz[0];
    k_zero_h<<<eblocks(P * 64), EL_THREADS, 0, stream>>>(packedIn, P * 64);
    k_pack_f32<<<eblocks(P), EL_THREADS, 0, stream>>>(flow2raw, 1, packedIn, 64, 0, P);
    k_pack_f16<<<eblocks(P * 32), EL_THREADS, 0, stream>>>(o5, 32, packedIn, 64, 1, P * 32);
    conv(packedIn, 64, nullptr, 0, cn[0], o1, 128, nullptr, H, W, 1, 1);
    conv(o1, 128, nullptr, 0, cn[1], o2, 128, nullptr, H, W, 2, 1);
    conv(o2, 128, nullptr, 0, cn[2], o1, 128, nullptr, H, W, 4, 1);
    conv(o1, 128, nullptr, 0, cn[3], o2, 96, nullptr, H, W, 8, 1);
    conv(o2, 96, nullptr, 0, cn[4], o1, 64, nullptr, H, W, 16, 1);
    conv(o1, 64, nullptr, 0, cn[5], o2, 32, nullptr, H, W, 1, 1);
    conv(o2, 32, nullptr, 0, cn[6], nullptr, 0, cnOut, H, W, 1, 0);
    k_relu_add<<<eblocks(P), EL_THREADS, 0, stream>>>(cnOut, flow2raw, flow2, P);
  }

  // ---- disparity outputs: disp = flow_x / W_level ----
  float* out = (float*)d_out;
  k_scale<<<eblocks(Ps[0]), EL_THREADS, 0, stream>>>(flow2, out, 1.0f / 208.0f, Ps[0]);
  k_scale<<<eblocks(Ps[1]), EL_THREADS, 0, stream>>>(flowB[1], out + Ps[0], 1.0f / 104.0f, Ps[1]);
  k_scale<<<eblocks(Ps[2]), EL_THREADS, 0, stream>>>(flowB[2], out + Ps[0] + Ps[1], 1.0f / 52.0f, Ps[2]);
  k_scale<<<eblocks(Ps[3]), EL_THREADS, 0, stream>>>(flowB[3], out + Ps[0] + Ps[1] + Ps[2], 1.0f / 26.0f, Ps[3]);
}
